// FraudHeteroGNN_39376260169908
// MI455X (gfx1250) — matrix operand order repaired and checked by reading, not compiled, not run on hardware
//
#include <hip/hip_runtime.h>
#include <hip/hip_bf16.h>
#include <math.h>

typedef __bf16 bf16;
typedef __attribute__((ext_vector_type(16))) __bf16 v16bf;
typedef __attribute__((ext_vector_type(8)))  __bf16 v8bf;
typedef __attribute__((ext_vector_type(8)))  float  v8f;

#define HID 128

// ---------------------------------------------------------------------------
// utility kernels
// ---------------------------------------------------------------------------
__global__ void k_cast_bf16(const float* __restrict__ s, bf16* __restrict__ d, size_t n) {
  size_t i = (size_t)blockIdx.x * blockDim.x + threadIdx.x;
  size_t st = (size_t)gridDim.x * blockDim.x;
  for (; i < n; i += st) d[i] = (bf16)s[i];
}

__global__ void k_fill_f32(float* __restrict__ p, float v, size_t n) {
  size_t i = (size_t)blockIdx.x * blockDim.x + threadIdx.x;
  size_t st = (size_t)gridDim.x * blockDim.x;
  for (; i < n; i += st) p[i] = v;
}

__global__ void k_rowbias(float* __restrict__ agg, const float* __restrict__ b, size_t total) {
  size_t i = (size_t)blockIdx.x * blockDim.x + threadIdx.x;
  size_t st = (size_t)gridDim.x * blockDim.x;
  for (; i < total; i += st) agg[i] += b[i & (HID - 1)];
}

// ---------------------------------------------------------------------------
// Pack weight matrix W[K,N] (f32, row-major) into WMMA B-fragment order:
//   dst[ ((kstep*ntiles + ntile)*32 + lane)*16 + j ]
//     = W[ kstep*32 + (lane>>4)*16 + j ][ ntile*16 + (lane&15) ]
// so each lane's 16 B-elements for one 16x16 fragment are 32 contiguous bytes.
// ---------------------------------------------------------------------------
__global__ void k_pack_b(const float* __restrict__ src, bf16* __restrict__ dst, int K, int N) {
  int i = blockIdx.x * blockDim.x + threadIdx.x;
  if (i >= K * N) return;
  int j = i & 15;
  int lane = (i >> 4) & 31;
  int ntiles = N >> 4;
  int ntile = (i >> 9) % ntiles;
  int ks = i / (512 * ntiles);
  int kk = ks * 32 + (lane >> 4) * 16 + j;
  int col = ntile * 16 + (lane & 15);
  dst[i] = (bf16)src[(size_t)kk * N + col];
}

// ---------------------------------------------------------------------------
// WMMA bf16 GEMM: C[M,N] = A[M,K] @ B[K,N] (+ bias[N])
// A bf16 row-major, B fragment-packed bf16 (k_pack_b), C f32.
// Block = 256 threads = 8 waves stacked along M. Wave tile 32x64 =
// 2x4 fragments of 16x16, BK = 32 -> 8 v_wmma per K-step. No LDS:
// A fragments are two contiguous b128 loads/lane; B fragments one
// contiguous 32B load/lane from the packed layout.
// ---------------------------------------------------------------------------
union AFrag { v16bf v; v8bf h[2]; };

__global__ __launch_bounds__(256)
void k_gemm_bias_wmma(const bf16* __restrict__ A, const bf16* __restrict__ Bp,
                      const float* __restrict__ bias, float* __restrict__ C,
                      int M, int K, int N) {
  const int tid = threadIdx.x;
  const int wave = tid >> 5;
  const int lane = tid & 31;
  const int rowBase = blockIdx.x * 256 + wave * 32;  // wave's 32 rows
  const int colBase = blockIdx.y * 64;               // 64 cols
  const int r    = lane & 15;
  const int hsel = lane >> 4;
  const int akb  = hsel * 16;       // A K-offset within step: lanes<16 -> 0, >=16 -> 16
  const int ntiles = N >> 4;
  const int nt0 = colBase >> 4;

  const v8f vz = {0.f, 0.f, 0.f, 0.f, 0.f, 0.f, 0.f, 0.f};
  v8f acc[2][4];
#pragma unroll
  for (int mi = 0; mi < 2; ++mi)
#pragma unroll
    for (int ni = 0; ni < 4; ++ni) acc[mi][ni] = vz;

  // clamp A rows at the M edge; out-of-range output rows are never stored
  int arow[2];
#pragma unroll
  for (int mi = 0; mi < 2; ++mi) {
    int rr = rowBase + mi * 16 + r;
    arow[mi] = (rr < M) ? rr : (M - 1);
  }

  const int ksteps = K >> 5;
  for (int ks = 0; ks < ksteps; ++ks) {
    const int k0 = ks << 5;
    // B fragments: contiguous 32B per lane from packed layout
    AFrag bfrag[4];
#pragma unroll
    for (int ni = 0; ni < 4; ++ni) {
      const bf16* bp = Bp + (((size_t)(ks * ntiles + nt0 + ni) * 32 + lane) << 4);
      bfrag[ni].h[0] = *(const v8bf*)bp;
      bfrag[ni].h[1] = *(const v8bf*)(bp + 8);
    }
#pragma unroll
    for (int mi = 0; mi < 2; ++mi) {
      const bf16* ap = A + (size_t)arow[mi] * K + k0 + akb;
      AFrag af;
      af.h[0] = *(const v8bf*)ap;         // K = k0+akb .. +7
      af.h[1] = *(const v8bf*)(ap + 8);   // K = k0+akb+8 .. +15
      if (ks + 1 < ksteps) __builtin_prefetch(ap + 32, 0, 0);
#pragma unroll
      for (int ni = 0; ni < 4; ++ni)
        acc[mi][ni] = __builtin_amdgcn_wmma_f32_16x16x32_bf16(
            false, af.v, false, bfrag[ni].v, (short)0, acc[mi][ni], false, false);
    }
  }

  const bool full = (rowBase + 32) <= M;
#pragma unroll
  for (int mi = 0; mi < 2; ++mi) {
#pragma unroll
    for (int ni = 0; ni < 4; ++ni) {
      const int ccol = colBase + ni * 16 + r;
      const float bv = bias ? bias[ccol] : 0.f;
      const int crow0 = rowBase + mi * 16 + hsel * 8;
      if (full) {
#pragma unroll
        for (int rr = 0; rr < 8; ++rr)
          C[(size_t)(crow0 + rr) * N + ccol] = acc[mi][ni][rr] + bv;
      } else {
#pragma unroll
        for (int rr = 0; rr < 8; ++rr)
          if (crow0 + rr < M) C[(size_t)(crow0 + rr) * N + ccol] = acc[mi][ni][rr] + bv;
      }
    }
  }
}

// ---------------------------------------------------------------------------
// LayerNorm (+ optional residual) + ELU. One row per block, blockDim == N.
// ---------------------------------------------------------------------------
__global__ void k_ln_elu(const float* __restrict__ x, const float* __restrict__ res,
                         const float* __restrict__ g, const float* __restrict__ beta,
                         float* __restrict__ out, int N) {
  const int row = blockIdx.x;
  const int t = threadIdx.x;
  const size_t idx = (size_t)row * N + t;
  float v = x[idx];
  if (res) v += res[idx];

  __shared__ float red[4];
  const int nw = blockDim.x >> 5;
  const int wid = t >> 5, lane = t & 31;

  float s = v;
  for (int o = 16; o > 0; o >>= 1) s += __shfl_xor(s, o, 32);
  if (lane == 0) red[wid] = s;
  __syncthreads();
  float tot = 0.f;
  for (int i = 0; i < nw; ++i) tot += red[i];
  const float mean = tot / N;
  __syncthreads();

  const float d = v - mean;
  float s2 = d * d;
  for (int o = 16; o > 0; o >>= 1) s2 += __shfl_xor(s2, o, 32);
  if (lane == 0) red[wid] = s2;
  __syncthreads();
  float vtot = 0.f;
  for (int i = 0; i < nw; ++i) vtot += red[i];
  const float var = vtot / N;

  float y = d * rsqrtf(var + 1e-5f) * g[t] + beta[t];
  out[idx] = (y > 0.f) ? y : expm1f(y);
}

// ---------------------------------------------------------------------------
// attention scores: out[node,h] = sum_d hs[node, h*32+d] * a[h*32+d]
// ---------------------------------------------------------------------------
__global__ void k_att(const float* __restrict__ hs, const float* __restrict__ a,
                      float* __restrict__ out, int n) {
  int i = blockIdx.x * blockDim.x + threadIdx.x;
  if (i >= n * 4) return;
  int node = i >> 2, h = i & 3;
  const float* hp = hs + (size_t)node * HID + h * 32;
  const float* ap = a + h * 32;
  float s = 0.f;
#pragma unroll 8
  for (int d = 0; d < 32; ++d) s += hp[d] * ap[d];
  out[i] = s;
}

// ---------------------------------------------------------------------------
// edge segment-softmax passes
// ---------------------------------------------------------------------------
__device__ inline void atomicMaxF(float* addr, float val) {
  unsigned* ua = (unsigned*)addr;
  unsigned old = *ua;
  while (true) {
    float f = __uint_as_float(old);
    if (f >= val) break;
    unsigned assumed = old;
    old = atomicCAS(ua, assumed, __float_as_uint(val));
    if (old == assumed) break;
  }
}

__device__ inline float leaky(float e) { return e > 0.f ? e : 0.2f * e; }

__global__ void k_edge_max(const int* __restrict__ src, const int* __restrict__ dst,
                           const float* __restrict__ as_, const float* __restrict__ ad_,
                           float* __restrict__ m, int E) {
  int i = blockIdx.x * blockDim.x + threadIdx.x;
  if (i >= E * 4) return;
  int e = i >> 2, h = i & 3;
  int s = src[e], d = dst[e];
  float ev = leaky(as_[s * 4 + h] + ad_[d * 4 + h]);
  atomicMaxF(&m[d * 4 + h], ev);
}

__global__ void k_edge_expsum(const int* __restrict__ src, const int* __restrict__ dst,
                              const float* __restrict__ as_, const float* __restrict__ ad_,
                              const float* __restrict__ m, float* __restrict__ den, int E) {
  int i = blockIdx.x * blockDim.x + threadIdx.x;
  if (i >= E * 4) return;
  int e = i >> 2, h = i & 3;
  int s = src[e], d = dst[e];
  float ev = leaky(as_[s * 4 + h] + ad_[d * 4 + h]);
  atomicAdd(&den[d * 4 + h], expf(ev - m[d * 4 + h]));
}

// one edge per 128-thread block; alpha computed once in LDS
__global__ __launch_bounds__(128)
void k_edge_scatter(const int* __restrict__ src, const int* __restrict__ dst,
                    const float* __restrict__ as_, const float* __restrict__ ad_,
                    const float* __restrict__ m, const float* __restrict__ den,
                    const float* __restrict__ hs, float* __restrict__ agg, int E) {
  __shared__ float alpha[4];
  int e = blockIdx.x;
  if (e >= E) return;
  int t = threadIdx.x;
  int s = src[e], d = dst[e];
  if (t < 4) {
    float ev = leaky(as_[s * 4 + t] + ad_[d * 4 + t]);
    float w = expf(ev - m[d * 4 + t]);
    alpha[t] = w / (den[d * 4 + t] + 1e-16f);
  }
  __syncthreads();
  float msg = hs[(size_t)s * HID + t] * alpha[t >> 5];
  atomicAdd(&agg[(size_t)d * HID + t], msg);
}

// ---------------------------------------------------------------------------
// fused classifier: relu(emb@W1 + b1) @ W2 + b2 ; one wave per row
// ---------------------------------------------------------------------------
__global__ __launch_bounds__(256)
void k_cls(const float* __restrict__ emb, const float* __restrict__ W1,
           const float* __restrict__ b1, const float* __restrict__ W2,
           const float* __restrict__ b2, float* __restrict__ out, int rows) {
  int gw = (blockIdx.x * blockDim.x + threadIdx.x) >> 5;
  int lane = threadIdx.x & 31;
  if (gw >= rows) return;
  const float* e = emb + (size_t)gw * 64;
  float hc = b1[lane];
#pragma unroll 8
  for (int k = 0; k < 64; ++k) hc += e[k] * W1[k * 32 + lane];
  hc = hc > 0.f ? hc : 0.f;
  float v = hc * W2[lane];
  for (int o = 16; o > 0; o >>= 1) v += __shfl_xor(v, o, 32);
  if (lane == 0) out[gw] = v + b2[0];
}

// ---------------------------------------------------------------------------
// host orchestration
// ---------------------------------------------------------------------------
extern "C" void kernel_launch(void* const* d_in, const int* in_sizes, int n_in,
                              void* d_out, int out_size, void* d_ws, size_t ws_size,
                              hipStream_t stream) {
  const int NT = in_sizes[0] / 256;
  const int NC = in_sizes[1] / 32;
  const int NA = in_sizes[2] / 32;
  const int nNode[3] = {NT, NC, NA};

  auto F = [&](int i) -> const float* { return (const float*)d_in[i]; };

  // ---- workspace layout ----
  char* w = (char*)d_ws;
  size_t off = 0;
  auto alloc = [&](size_t bytes) -> char* {
    char* p = w + off;
    off += (bytes + 255) & ~(size_t)255;
    return p;
  };
  float* h[3];   h[0]   = (float*)alloc((size_t)NT * HID * 4);
                 h[1]   = (float*)alloc((size_t)NC * HID * 4);
                 h[2]   = (float*)alloc((size_t)NA * HID * 4);
  bf16*  hbf[3]; hbf[0] = (bf16*)alloc((size_t)NT * HID * 2);
                 hbf[1] = (bf16*)alloc((size_t)NC * HID * 2);
                 hbf[2] = (bf16*)alloc((size_t)NA * HID * 2);
  float* agg[3]; agg[0] = (float*)alloc((size_t)NT * HID * 4);
                 agg[1] = (float*)alloc((size_t)NC * HID * 4);
                 agg[2] = (float*)alloc((size_t)NA * HID * 4);
  // scr0: x_tx as bf16 (NT*256*2B) OR hs f32 (NT*128*4B) -- same size
  float* scr0 = (float*)alloc((size_t)NT * 256 * 2);
  // scr1: x_card/addr bf16, hd f32, emb f32 -- max NT*128*4
  float* scr1 = (float*)alloc((size_t)NT * HID * 4);
  float* a_s  = (float*)alloc((size_t)NT * 4 * 4);
  float* a_d  = (float*)alloc((size_t)NT * 4 * 4);
  float* mbuf = (float*)alloc((size_t)NT * 4 * 4);
  float* dbuf = (float*)alloc((size_t)NT * 4 * 4);
  bf16* wbf = (bf16*)alloc((size_t)442368 * 2);
  if (off > ws_size) return;  // workspace too small: bail (deterministically)

  // ---- launch helpers ----
  auto cast = [&](const float* s, bf16* d, size_t n) {
    int blocks = (int)((n + 255) / 256); if (blocks > 65535) blocks = 65535;
    k_cast_bf16<<<blocks, 256, 0, stream>>>(s, d, n);
  };
  auto packB = [&](const float* s, bf16* d, int K, int N) {
    int blocks = (K * N + 255) / 256;
    k_pack_b<<<blocks, 256, 0, stream>>>(s, d, K, N);
  };
  auto fill = [&](float* p, float v, size_t n) {
    int blocks = (int)((n + 255) / 256); if (blocks > 65535) blocks = 65535;
    k_fill_f32<<<blocks, 256, 0, stream>>>(p, v, n);
  };
  auto gemm = [&](const bf16* A, const bf16* B, const float* bias, float* C,
                  int M, int K, int N) {
    dim3 g((M + 255) / 256, N / 64);
    k_gemm_bias_wmma<<<g, 256, 0, stream>>>(A, B, bias, C, M, K, N);
  };
  auto lnelu = [&](const float* x, const float* res, const float* g_, const float* b_,
                   float* o, int rows, int N) {
    k_ln_elu<<<rows, N, 0, stream>>>(x, res, g_, b_, o, N);
  };

  // ---- param leaf indices (jax pytree: dict keys sorted) ----
  // 0..2 x_*, 3..6 ei_*,
  // cls: 7 W1, 8 W2, 9 b1, 10 b2
  // emb: 11 W, 12 b, 13 beta, 14 g
  // layers[l] base = 15 + 26*l:
  //   conv (sorted addr_tx, card_tx, tx_addr, tx_card), 5 leaves each: Wd,Ws,ad,as,b
  //   ln   (sorted address, card, transaction), 2 leaves each: beta,g
  // proj base 93: address(W,b,beta,g), card(97..), transaction(101..)

  // weight bf16 region layout (element offsets; all fragment-packed)
  const size_t WPROJ_TX = 0, WPROJ_CARD = 32768, WPROJ_ADDR = 36864;
  const size_t WLAYER0 = 40960;                 // (l*4+slot)*32768 : Ws then Wd
  const size_t WEMB = 434176;

  // ---- pack all weights to fragment-ordered bf16 ----
  packB(F(101), wbf + WPROJ_TX,   256, HID);
  packB(F(97),  wbf + WPROJ_CARD, 32, HID);
  packB(F(93),  wbf + WPROJ_ADDR, 32, HID);
  for (int l = 0; l < 3; ++l)
    for (int slot = 0; slot < 4; ++slot) {
      int base = 15 + 26 * l + 5 * slot;
      size_t o2 = WLAYER0 + (size_t)(l * 4 + slot) * 32768;
      packB(F(base + 1), wbf + o2,         HID, HID);  // Ws
      packB(F(base + 0), wbf + o2 + 16384, HID, HID);  // Wd
    }
  packB(F(11), wbf + WEMB, HID, 64);

  // ---- input projections ----
  cast((const float*)d_in[0], (bf16*)scr0, (size_t)NT * 256);
  gemm((const bf16*)scr0, wbf + WPROJ_TX, F(102), h[0], NT, 256, HID);
  lnelu(h[0], nullptr, F(104), F(103), h[0], NT, HID);

  cast((const float*)d_in[1], (bf16*)scr1, (size_t)NC * 32);
  gemm((const bf16*)scr1, wbf + WPROJ_CARD, F(98), h[1], NC, 32, HID);
  lnelu(h[1], nullptr, F(100), F(99), h[1], NC, HID);

  cast((const float*)d_in[2], (bf16*)scr1, (size_t)NA * 32);
  gemm((const bf16*)scr1, wbf + WPROJ_ADDR, F(94), h[2], NA, 32, HID);
  lnelu(h[2], nullptr, F(96), F(95), h[2], NA, HID);

  // ---- edge type table: {ei input idx, conv slot, src type, dst type} ----
  struct ET { int ei, slot, s, d; };
  const ET ets[4] = {
    {3, 3, 0, 1},  // tx_card
    {4, 1, 1, 0},  // card_tx
    {5, 2, 0, 2},  // tx_addr
    {6, 0, 2, 0},  // addr_tx
  };

  // ---- GAT layers ----
  for (int l = 0; l < 3; ++l) {
    for (int t = 0; t < 3; ++t) {
      cast(h[t], hbf[t], (size_t)nNode[t] * HID);
      fill(agg[t], 0.f, (size_t)nNode[t] * HID);
    }
    for (int k = 0; k < 4; ++k) {
      const ET& et = ets[k];
      const int Ns = nNode[et.s], Nd = nNode[et.d];
      const int* ei = (const int*)d_in[et.ei];
      const int E = in_sizes[et.ei] / 2;
      const int* esrc = ei;
      const int* edst = ei + E;
      const int base = 15 + 26 * l + 5 * et.slot;
      const bf16* Wsbf = wbf + WLAYER0 + (size_t)(l * 4 + et.slot) * 32768;
      const bf16* Wdbf = Wsbf + 16384;
      const float* ad = F(base + 2);
      const float* as = F(base + 3);
      const float* bb = F(base + 4);

      gemm(hbf[et.s], Wsbf, nullptr, scr0, Ns, HID, HID);  // hs
      gemm(hbf[et.d], Wdbf, nullptr, scr1, Nd, HID, HID);  // hd
      k_att<<<(Ns * 4 + 255) / 256, 256, 0, stream>>>(scr0, as, a_s, Ns);
      k_att<<<(Nd * 4 + 255) / 256, 256, 0, stream>>>(scr1, ad, a_d, Nd);

      fill(mbuf, -3.0e38f, (size_t)Nd * 4);
      k_edge_max<<<(E * 4 + 255) / 256, 256, 0, stream>>>(esrc, edst, a_s, a_d, mbuf, E);
      fill(dbuf, 0.f, (size_t)Nd * 4);
      k_edge_expsum<<<(E * 4 + 255) / 256, 256, 0, stream>>>(esrc, edst, a_s, a_d, mbuf, dbuf, E);
      k_edge_scatter<<<E, 128, 0, stream>>>(esrc, edst, a_s, a_d, mbuf, dbuf, scr0, agg[et.d], E);
      {
        size_t total = (size_t)Nd * HID;
        int blocks = (int)((total + 255) / 256); if (blocks > 65535) blocks = 65535;
        k_rowbias<<<blocks, 256, 0, stream>>>(agg[et.d], bb, total);
      }
    }
    // residual + LN + ELU, in place into h
    const int lnBeta[3] = {15 + 26 * l + 24, 15 + 26 * l + 22, 15 + 26 * l + 20};
    for (int t = 0; t < 3; ++t)
      lnelu(agg[t], h[t], F(lnBeta[t] + 1), F(lnBeta[t]), h[t], nNode[t], HID);
  }

  // ---- embedding + classifier ----
  cast(h[0], hbf[0], (size_t)NT * HID);
  gemm(hbf[0], wbf + WEMB, F(12), scr1, NT, HID, 64);
  lnelu(scr1, nullptr, F(14), F(13), scr1, NT, 64);
  k_cls<<<(NT + 7) / 8, 256, 0, stream>>>(scr1, F(7), F(9), F(8), F(10), (float*)d_out, NT);
}
